// Linformer_9947144258098
// MI455X (gfx1250) — matrix operand order repaired and checked by reading, not compiled, not run on hardware
//
#include <hip/hip_runtime.h>
#include <hip/hip_bf16.h>

typedef __attribute__((ext_vector_type(16))) __bf16 v16bf;
typedef __attribute__((ext_vector_type(8)))  float  v8f;
typedef __attribute__((ext_vector_type(2)))  __bf16 bf16x2;

#define B_  4
#define H_  16
#define S_  4096
#define D_  64
#define R_  256

__device__ __forceinline__ __bf16 f2bf(float x) { return (__bf16)x; }

// 32-bit LDS offset of a __shared__ object (addrspace(3) pointer value).
__device__ __forceinline__ unsigned lds_off(const void* p) {
    return (unsigned)(unsigned long long)
        (__attribute__((address_space(3))) const char*)p;
}

// A-fragment K index for 16-bit 16x32 A tiles (contiguous, matches B):
// element e in lane L (half = L>>4): k = half*16 + e
__device__ __forceinline__ int a_kidx(int p /*pair*/, int half) {
    return (half << 4) + 2 * p;
}

// ---------------------------------------------------------------------------
// Kernel 1: kp = (R/sqrt(256)) @ (mask*K),  vp = (R/sqrt(256)) @ (mask*V)
// kp stored [bh][r][d] bf16 ; vp stored transposed [bh][d][r] bf16.
// grid = B*H*4 (4 r-blocks of 64 rows), block = 256 threads (8 waves).
// Ping-pong LDS double buffering: one barrier per k-step, global prefetch
// overlaps the WMMA phase.
// ---------------------------------------------------------------------------
__global__ __launch_bounds__(256) void linf_project(
    const float* __restrict__ key, const float* __restrict__ val,
    const float* __restrict__ mask, const float* __restrict__ Rm,
    __bf16* __restrict__ kp, __bf16* __restrict__ vpT)
{
    __shared__ __align__(16) __bf16 ldsR [2][64][32];   // Rs tile [r_local][s_local]
    __shared__ __align__(16) __bf16 ldsKT[2][64][32];   // K tile transposed [d][s_local]
    __shared__ __align__(16) __bf16 ldsVT[2][64][32];   // V tile transposed [d][s_local]

    const int bh = blockIdx.x >> 2;     // 0..63
    const int rb = blockIdx.x & 3;      // r-block
    const int b  = bh >> 4;             // batch (H=16)
    const int r0 = rb << 6;             // 64 r rows per block

    const int t    = threadIdx.x;
    const int lane = t & 31;
    const int w    = t >> 5;            // wave 0..7
    const int half = lane >> 4;
    const int ln   = lane & 15;
    const int rt   = w >> 1;            // r-tile 0..3 (16 rows each)

    const float* Kb = key + (size_t)bh * S_ * D_;
    const float* Vb = val + (size_t)bh * S_ * D_;
    const float* Mb = mask + (size_t)b * S_;

    v8f acc[4] = {};                    // 4 d-tiles of 16x16 f32

    const int ri  = t >> 2;             // staging: R row 0..63
    const int scl = (t & 3) << 3;       // staging: R col base (8 floats)
    const int si  = t >> 3;             // staging: K/V row 0..31
    const int dc  = (t & 7) << 3;       // staging: K/V col base (8 floats)

    // prefetch registers
    float4 rA, rB, kA, kB, vA, vB;
    float  mval;

    auto load_tile = [&](int s0) {
        const float* rs = Rm + (size_t)(r0 + ri) * S_ + (s0 + scl);
        rA = ((const float4*)rs)[0];
        rB = ((const float4*)rs)[1];
        mval = Mb[s0 + si];
        const float* ks = Kb + (size_t)(s0 + si) * D_ + dc;
        const float* vs = Vb + (size_t)(s0 + si) * D_ + dc;
        kA = ((const float4*)ks)[0];
        kB = ((const float4*)ks)[1];
        vA = ((const float4*)vs)[0];
        vB = ((const float4*)vs)[1];
    };
    auto store_tile = [&](int buf) {
        const float q = 0.0625f;        // 1/sqrt(RANK)
        bf16x2* dst = (bf16x2*)&ldsR[buf][ri][scl];
        bf16x2 p0; p0.x = f2bf(rA.x * q); p0.y = f2bf(rA.y * q);
        bf16x2 p1; p1.x = f2bf(rA.z * q); p1.y = f2bf(rA.w * q);
        bf16x2 p2; p2.x = f2bf(rB.x * q); p2.y = f2bf(rB.y * q);
        bf16x2 p3; p3.x = f2bf(rB.z * q); p3.y = f2bf(rB.w * q);
        dst[0] = p0; dst[1] = p1; dst[2] = p2; dst[3] = p3;
        ldsKT[buf][dc+0][si] = f2bf(kA.x*mval); ldsKT[buf][dc+1][si] = f2bf(kA.y*mval);
        ldsKT[buf][dc+2][si] = f2bf(kA.z*mval); ldsKT[buf][dc+3][si] = f2bf(kA.w*mval);
        ldsKT[buf][dc+4][si] = f2bf(kB.x*mval); ldsKT[buf][dc+5][si] = f2bf(kB.y*mval);
        ldsKT[buf][dc+6][si] = f2bf(kB.z*mval); ldsKT[buf][dc+7][si] = f2bf(kB.w*mval);
        ldsVT[buf][dc+0][si] = f2bf(vA.x*mval); ldsVT[buf][dc+1][si] = f2bf(vA.y*mval);
        ldsVT[buf][dc+2][si] = f2bf(vA.z*mval); ldsVT[buf][dc+3][si] = f2bf(vA.w*mval);
        ldsVT[buf][dc+4][si] = f2bf(vB.x*mval); ldsVT[buf][dc+5][si] = f2bf(vB.y*mval);
        ldsVT[buf][dc+6][si] = f2bf(vB.z*mval); ldsVT[buf][dc+7][si] = f2bf(vB.w*mval);
    };

    load_tile(0);
    store_tile(0);

    for (int it = 0; it < 128; ++it) {
        const int cur = it & 1;
        if (it < 127) load_tile((it + 1) << 5);   // prefetch next (long latency)
        __syncthreads();                           // buf[cur] ready, buf[next] free

        // ---- A fragment: Rs rows rt*16+ln, K = s-local, contiguous k = half*16 + e ----
        v16bf a;
        {
            const __bf16* row = &ldsR[cur][(rt << 4) + ln][0];
            #pragma unroll
            for (int p = 0; p < 8; ++p) {
                bf16x2 pr = *(const bf16x2*)(row + a_kidx(p, half));
                a[2*p] = pr.x; a[2*p+1] = pr.y;
            }
        }
        // ---- B fragments + WMMA for 4 d-tiles ----
        const __bf16 (*bt)[32] = (w & 1) ? ldsVT[cur] : ldsKT[cur];
        #pragma unroll
        for (int dt = 0; dt < 4; ++dt) {
            v16bf bmat;
            const __bf16* col = &bt[(dt << 4) + ln][0];   // N = dt*16+ln
            #pragma unroll
            for (int p = 0; p < 8; ++p) {                 // K = half*16 + e
                bf16x2 pr = *(const bf16x2*)(col + (half << 4) + 2*p);
                bmat[2*p] = pr.x; bmat[2*p+1] = pr.y;
            }
            acc[dt] = __builtin_amdgcn_wmma_f32_16x16x32_bf16(
                false, a, false, bmat, (short)0, acc[dt], false, false);
        }
        if (it < 127) store_tile((it + 1) & 1);   // fill the other buffer
    }

    __syncthreads();    // main loop done; LDS reusable for kp epilogue bounce

    // ---- epilogue: C layout M = v + 8*half, N = ln ----
    const int rbase = rt << 4;   // r_local base within this 64-row block
    __bf16 (*ldsOut)[64] = reinterpret_cast<__bf16(*)[64]>(&ldsR[0][0][0]); // 8 KB

    if ((w & 1) == 0) {
        // kp tiles -> LDS bounce (scattered stores go to LDS, not HBM)
        #pragma unroll
        for (int dt = 0; dt < 4; ++dt)
            #pragma unroll
            for (int v = 0; v < 8; ++v)
                ldsOut[rbase + v + (half << 3)][(dt << 4) + ln] = f2bf(acc[dt][v]);
    } else {
        // vpT[bh][d][r]: r consecutive per lane -> vectorizes to b128 stores
        __bf16* out = vpT + (size_t)bh * D_ * R_;
        #pragma unroll
        for (int dt = 0; dt < 4; ++dt)
            #pragma unroll
            for (int v = 0; v < 8; ++v) {
                const int r = r0 + rbase + v + (half << 3);
                const int d = (dt << 4) + ln;
                out[(size_t)d * R_ + r] = f2bf(acc[dt][v]);
            }
    }
    __syncthreads();

    // cooperative, fully coalesced kp store: loop copies 1024 uint4 (16 KB), twice the 512-uint4 64x64 bf16 tile
    {
        uint4* gdst = (uint4*)(kp + (size_t)bh * R_ * D_ + (size_t)r0 * D_);
        const uint4* lsrc = (const uint4*)&ldsOut[0][0];
        #pragma unroll
        for (int i = 0; i < 4; ++i)
            gdst[t + 256 * i] = lsrc[t + 256 * i];
    }
}

// ---------------------------------------------------------------------------
// Kernel 2: scores = Q@kp^T / 8 ; softmax over r ; out = probs@vp
// grid = B*H * (S/128), block = 256 (8 waves, 16 query rows per wave).
// ---------------------------------------------------------------------------
__global__ __launch_bounds__(256) void linf_attend(
    const float* __restrict__ query, const __bf16* __restrict__ kp,
    const __bf16* __restrict__ vpT, float* __restrict__ out)
{
    __shared__ __align__(16) __bf16 ldsKP[R_][D_];     // 32 KB, kp[r][d]
    __shared__ __align__(16) __bf16 ldsP[8][16][32];   // 1 KB/wave probs chunk

    const int bh = blockIdx.x >> 5;   // 32 s-blocks per head
    const int sb = blockIdx.x & 31;
    const int t = threadIdx.x, lane = t & 31, w = t >> 5;
    const int half = lane >> 4, ln = lane & 15;

    // async copy of kp[bh] into LDS: 2048 x 16B, ASYNCcnt-tracked, no VGPR bounce
    {
        const uint4* src = (const uint4*)(kp + (size_t)bh * R_ * D_);
        const unsigned base = lds_off(&ldsKP[0][0]);
        #pragma unroll
        for (int i = 0; i < 8; ++i) {
            unsigned dst_lds = base + (unsigned)(t + 256 * i) * 16u;
            unsigned long long ga = (unsigned long long)(src + (t + 256 * i));
            asm volatile("global_load_async_to_lds_b128 %0, %1, off"
                         :: "v"(dst_lds), "v"(ga) : "memory");
        }
        asm volatile("s_wait_asynccnt 0x0" ::: "memory");
    }
    __syncthreads();

    const int s_base = (sb << 7) + (w << 4);
    const float* Q = query + (size_t)bh * S_ * D_;

    // Q A-fragments (two 16x32 k-steps over d)
    v16bf qa[2];
    #pragma unroll
    for (int kd = 0; kd < 2; ++kd) {
        const float* qrow = Q + (size_t)(s_base + ln) * D_ + (kd << 5);
        #pragma unroll
        for (int p = 0; p < 8; ++p) {
            float2 f = *(const float2*)(qrow + a_kidx(p, half));
            qa[kd][2*p] = f2bf(f.x); qa[kd][2*p+1] = f2bf(f.y);
        }
    }

    // scores strip: 16 r-tiles of 16x16 f32
    v8f sc[16];
    #pragma unroll
    for (int n = 0; n < 16; ++n) {
        v8f acc = {};
        #pragma unroll
        for (int kd = 0; kd < 2; ++kd) {
            v16bf bmat;
            const __bf16* row = &ldsKP[(n << 4) + ln][0];   // N = r = n*16+ln
            #pragma unroll
            for (int p = 0; p < 8; ++p) {                   // K = d
                bf16x2 pr = *(const bf16x2*)(row + (kd << 5) + (half << 4) + 2*p);
                bmat[2*p] = pr.x; bmat[2*p+1] = pr.y;
            }
            acc = __builtin_amdgcn_wmma_f32_16x16x32_bf16(
                false, qa[kd], false, bmat, (short)0, acc, false, false);
        }
        sc[n] = acc;
    }

    // softmax per row s = 8*half + v : reduce over n (regs) then 16 lanes of half
    const float scale = 0.125f;  // 1/sqrt(D)
    float rinv[8];
    #pragma unroll
    for (int v = 0; v < 8; ++v) {
        float m = -3.4e38f;
        #pragma unroll
        for (int n = 0; n < 16; ++n) m = fmaxf(m, sc[n][v]);
        m = fmaxf(m, __shfl_xor(m, 1, 32));
        m = fmaxf(m, __shfl_xor(m, 2, 32));
        m = fmaxf(m, __shfl_xor(m, 4, 32));
        m = fmaxf(m, __shfl_xor(m, 8, 32));
        float s = 0.f;
        #pragma unroll
        for (int n = 0; n < 16; ++n) {
            float e = __expf((sc[n][v] - m) * scale);
            sc[n][v] = e; s += e;
        }
        s += __shfl_xor(s, 1, 32);
        s += __shfl_xor(s, 2, 32);
        s += __shfl_xor(s, 4, 32);
        s += __shfl_xor(s, 8, 32);
        rinv[v] = 1.0f / s;
    }

    // out = probs @ vp : 8 k-steps of 32 r, via per-wave LDS swizzle to A layout
    const __bf16* vbase = vpT + (size_t)bh * D_ * R_;
    v8f oacc[4] = {};
    for (int kr = 0; kr < 8; ++kr) {
        #pragma unroll
        for (int j = 0; j < 2; ++j) {
            const int n = kr * 2 + j;
            #pragma unroll
            for (int v = 0; v < 8; ++v)   // C layout -> [s_local][r_local]
                ldsP[w][(half << 3) + v][(j << 4) + ln] = f2bf(sc[n][v] * rinv[v]);
        }
        // wave-internal DS RAW: DS ops are in-order per wave; fence + wait
        asm volatile("s_wait_dscnt 0" ::: "memory");
        v16bf pa;
        {
            const __bf16* row = &ldsP[w][ln][0];            // M = s = ln
            #pragma unroll
            for (int p = 0; p < 8; ++p) {
                bf16x2 pr = *(const bf16x2*)(row + a_kidx(p, half));
                pa[2*p] = pr.x; pa[2*p+1] = pr.y;
            }
        }
        #pragma unroll
        for (int dt = 0; dt < 4; ++dt) {
            v16bf bmat;
            const __bf16* col = vbase + (size_t)((dt << 4) + ln) * R_; // N = d
            #pragma unroll
            for (int p = 0; p < 8; ++p) {                   // K = r
                bf16x2 pr = *(const bf16x2*)(col + (kr << 5) + (half << 4) + 2*p);
                bmat[2*p] = pr.x; bmat[2*p+1] = pr.y;
            }
            oacc[dt] = __builtin_amdgcn_wmma_f32_16x16x32_bf16(
                false, pa, false, bmat, (short)0, oacc[dt], false, false);
        }
    }

    // store output f32: s = s_base + v + 8*half, d = dt*16 + ln
    float* O = out + (size_t)bh * S_ * D_;
    #pragma unroll
    for (int dt = 0; dt < 4; ++dt)
        #pragma unroll
        for (int v = 0; v < 8; ++v) {
            const int s = s_base + v + (half << 3);
            O[(size_t)s * D_ + (dt << 4) + ln] = oacc[dt][v];
        }
}

// ---------------------------------------------------------------------------
extern "C" void kernel_launch(void* const* d_in, const int* in_sizes, int n_in,
                              void* d_out, int out_size, void* d_ws, size_t ws_size,
                              hipStream_t stream) {
    (void)in_sizes; (void)n_in; (void)out_size; (void)ws_size;
    const float* q  = (const float*)d_in[0];
    const float* k  = (const float*)d_in[1];
    const float* v  = (const float*)d_in[2];
    const float* m  = (const float*)d_in[3];
    const float* Rm = (const float*)d_in[4];

    __bf16* kp  = (__bf16*)d_ws;                         // [64][256][64] bf16 (2 MiB)
    __bf16* vpT = kp + (size_t)B_ * H_ * R_ * D_;        // [64][64][256] bf16 (2 MiB)

    linf_project<<<B_ * H_ * 4, 256, 0, stream>>>(k, v, m, Rm, kp, vpT);
    linf_attend <<<B_ * H_ * (S_ / 128), 256, 0, stream>>>(q, kp, vpT, (float*)d_out);
}
